// _EDM_AttentionBlock_74552042324418
// MI455X (gfx1250) — matrix operand order repaired and checked by reading, not compiled, not run on hardware
//
#include <hip/hip_runtime.h>
#include <hip/hip_bf16.h>

typedef __attribute__((ext_vector_type(16))) _Float16 v16h;
typedef __attribute__((ext_vector_type(8)))  _Float16 v8h;
typedef __attribute__((ext_vector_type(4)))  _Float16 v4h;
typedef __attribute__((ext_vector_type(8)))  float    v8f;

#define BDIM_B   16
#define BDIM_C   512
#define BDIM_HW  4096
#define NHEADS   8
#define DHEAD    64
#define NGROUPS  32
#define GN_EPS   1e-6f

// ---------------------------------------------------------------------------
// Async global->LDS copy of 16B per lane (CDNA5 ASYNCcnt path).
// VDST operand carries the 32-bit LDS byte offset; VADDR is the 64-bit
// global address. Tracked by ASYNCcnt, drained with s_wait_asynccnt.
// ---------------------------------------------------------------------------
__device__ __forceinline__ void cp_async16(const _Float16* g, const _Float16* l) {
    unsigned loff = (unsigned)(size_t)(__attribute__((address_space(3))) const _Float16*)l;
    asm volatile("global_load_async_to_lds_b128 %0, %1, off"
                 :: "v"(loff), "v"(g) : "memory");
}
__device__ __forceinline__ void wait_async0() {
    asm volatile("s_wait_asynccnt 0x0" ::: "memory");
}

// ---------------------------------------------------------------------------
// Kernel 1: GroupNorm per (b, g); writes TRANSPOSED f16 activations
//           xn_t[b][hw][c]  (so GEMM B-tiles are K-contiguous, no transpose).
// ---------------------------------------------------------------------------
__global__ __launch_bounds__(256) void gn_kernel(const float* __restrict__ x,
                                                 const float* __restrict__ gamma,
                                                 const float* __restrict__ beta,
                                                 _Float16* __restrict__ xn_t) {
    const int g = blockIdx.x;                 // 0..31
    const int b = blockIdx.y;                 // 0..15
    const int CG = BDIM_C / NGROUPS;          // 16 channels per group
    const int NEL = CG * BDIM_HW;             // 65536 contiguous floats
    const size_t base = ((size_t)b * BDIM_C + (size_t)g * CG) * BDIM_HW;

    float sum = 0.f, sq = 0.f;
    for (int i = threadIdx.x; i < NEL; i += 256) {
        float v = x[base + i];
        sum += v; sq += v * v;
    }
    __shared__ float red[512];
    red[threadIdx.x] = sum;
    red[256 + threadIdx.x] = sq;
    __syncthreads();
    for (int off = 128; off > 0; off >>= 1) {
        if (threadIdx.x < off) {
            red[threadIdx.x]       += red[threadIdx.x + off];
            red[256 + threadIdx.x] += red[256 + threadIdx.x + off];
        }
        __syncthreads();
    }
    const float inv_n = 1.f / (float)NEL;
    const float mu  = red[0] * inv_n;
    const float var = red[256] * inv_n - mu * mu;
    const float rstd = rsqrtf(var + GN_EPS);

    float gscale[16], gshift[16];
#pragma unroll
    for (int cl = 0; cl < 16; ++cl) {
        const int c = g * CG + cl;
        gscale[cl] = rstd * gamma[c];
        gshift[cl] = beta[c] - mu * rstd * gamma[c];
    }

    // Normalize + transpose: each thread owns pixel p, gathers 16 channels
    // (coalesced strided loads), writes one contiguous 32B chunk.
    for (int p0 = 0; p0 < BDIM_HW; p0 += 256) {
        const int p = p0 + threadIdx.x;
        v8h o0, o1;
#pragma unroll
        for (int cl = 0; cl < 8; ++cl)
            o0[cl] = (_Float16)(x[base + (size_t)cl * BDIM_HW + p] * gscale[cl] + gshift[cl]);
#pragma unroll
        for (int cl = 8; cl < 16; ++cl)
            o1[cl - 8] = (_Float16)(x[base + (size_t)cl * BDIM_HW + p] * gscale[cl] + gshift[cl]);
        _Float16* dst = xn_t + ((size_t)b * BDIM_HW + p) * BDIM_C + g * CG;
        *(v8h*)dst       = o0;
        *(v8h*)(dst + 8) = o1;
    }
}

// ---------------------------------------------------------------------------
// Kernel 2: f32 -> f16 weight conversion (grid-stride)
// ---------------------------------------------------------------------------
__global__ __launch_bounds__(256) void cvt_kernel(const float* __restrict__ w,
                                                  _Float16* __restrict__ wh, int n) {
    for (int i = blockIdx.x * 256 + threadIdx.x; i < n; i += gridDim.x * 256)
        wh[i] = (_Float16)w[i];
}

// ---------------------------------------------------------------------------
// Kernel 3/5: WMMA GEMM  out[b][m][n] = sum_k W[m][k] * Act_t[b][n][k] (+bias)(+resid)
//   W: f16 row-major [Mtot][512]; Act_t: f16 [b][4096][512]  (K-contiguous!)
//   Block tile 128x128, BK=32, 256 threads = 8 waves, each wave 32x64.
//   Double-buffered LDS, filled by async global->LDS DMA overlapping WMMAs.
// ---------------------------------------------------------------------------
#define GK 512
#define GN 4096
#define LDA_S 48     // padded halves per LDS row (96B, 16B aligned)

template <bool PROJ>
__global__ __launch_bounds__(256) void gemm_kernel(const _Float16* __restrict__ W,
                                                   const _Float16* __restrict__ Act_t,
                                                   const float* __restrict__ bias,
                                                   const float* __restrict__ resid,
                                                   _Float16* __restrict__ out_h,
                                                   float* __restrict__ out_f,
                                                   int Mtot) {
    __shared__ _Float16 As[2][128 * LDA_S];   // [stage][m 0..127][k 0..31]
    __shared__ _Float16 Bt[2][128 * LDA_S];   // [stage][n 0..127][k 0..31]

    const int batch = blockIdx.z;
    const int m0 = blockIdx.y * 128;
    const int n0 = blockIdx.x * 128;
    const int tid  = threadIdx.x;
    const int lane = tid & 31;
    const int wave = tid >> 5;
    const int half = lane >> 4;       // 0/1
    const int l16  = lane & 15;
    const int wm = wave >> 1;         // 0..3  (32 rows each)
    const int wn = wave & 1;          // 0..1  (64 cols each)

    const _Float16* actt = Act_t + (size_t)batch * GN * GK;

    v8f acc[2][4];
#pragma unroll
    for (int mi = 0; mi < 2; ++mi)
#pragma unroll
        for (int ni = 0; ni < 4; ++ni)
            acc[mi][ni] = (v8f){};

    // fills: thread owns one 32B chunk of one row in each tile
    const int f_row = tid >> 1;            // 0..127
    const int f_col = (tid & 1) * 16;      // 0 / 16
    const int f_dst = f_row * LDA_S + f_col;

    // hoisted per-thread LDS fragment offsets
    int a_off[2];
#pragma unroll
    for (int mi = 0; mi < 2; ++mi)
        a_off[mi] = (wm * 32 + mi * 16 + l16) * LDA_S + half * 8;
    int b_off[4];
#pragma unroll
    for (int ni = 0; ni < 4; ++ni)
        b_off[ni] = (wn * 64 + ni * 16 + l16) * LDA_S + half * 8;

    const _Float16* a_src = W    + (size_t)(m0 + f_row) * GK + f_col;
    const _Float16* b_src = actt + (size_t)(n0 + f_row) * GK + f_col;

    // prologue: async-fill stage 0 (k=0)
    cp_async16(a_src,     &As[0][f_dst]);
    cp_async16(a_src + 8, &As[0][f_dst + 8]);
    cp_async16(b_src,     &Bt[0][f_dst]);
    cp_async16(b_src + 8, &Bt[0][f_dst + 8]);

    int s = 0;
    for (int k0 = 0; k0 < GK; k0 += 32, s ^= 1) {
        wait_async0();        // this wave's stage-s fills landed in LDS
        __syncthreads();      // all waves' fills landed; prior reads retired

        if (k0 + 32 < GK) {   // async-prefetch next stage under the WMMAs
            const int ns = s ^ 1;
            cp_async16(a_src + k0 + 32,     &As[ns][f_dst]);
            cp_async16(a_src + k0 + 40,     &As[ns][f_dst + 8]);
            cp_async16(b_src + k0 + 32,     &Bt[ns][f_dst]);
            cp_async16(b_src + k0 + 40,     &Bt[ns][f_dst + 8]);
        }

        // A fragments (16x32 f16): lane = row M; e0..7 -> K=half*8+e,
        // e8..15 -> K=16+half*8+(e-8)  => two contiguous 16B LDS loads.
        v16h afr[2];
#pragma unroll
        for (int mi = 0; mi < 2; ++mi) {
            const _Float16* ap = &As[s][a_off[mi]];
            ((v8h*)&afr[mi])[0] = *(const v8h*)ap;
            ((v8h*)&afr[mi])[1] = *(const v8h*)(ap + 16);
        }
        // B fragments (32x16 f16): lane = col N; e0..7 -> K=half*8+e,
        //   e8..15 -> K=16+half*8+(e-8)  => two 16B LDS loads.
        v16h bfr[4];
#pragma unroll
        for (int ni = 0; ni < 4; ++ni) {
            const _Float16* bp = &Bt[s][b_off[ni]];
            ((v8h*)&bfr[ni])[0] = *(const v8h*)bp;
            ((v8h*)&bfr[ni])[1] = *(const v8h*)(bp + 16);
        }
#pragma unroll
        for (int mi = 0; mi < 2; ++mi)
#pragma unroll
            for (int ni = 0; ni < 4; ++ni)
                acc[mi][ni] = __builtin_amdgcn_wmma_f32_16x16x32_f16(
                    false, afr[mi], false, bfr[ni],
                    (short)0, acc[mi][ni], false, false);
    }

    // Epilogue. C layout: VGPR r, lanes 0-15: M=r, N=lane; lanes 16-31: M=r+8.
#pragma unroll
    for (int mi = 0; mi < 2; ++mi) {
#pragma unroll
        for (int ni = 0; ni < 4; ++ni) {
            const int ng = n0 + wn * 64 + ni * 16 + l16;
#pragma unroll
            for (int r = 0; r < 8; ++r) {
                const int mg = m0 + wm * 32 + mi * 16 + r + 8 * half;
                float v = acc[mi][ni][r] + bias[mg];
                const size_t off = ((size_t)batch * Mtot + mg) * GN + ng;
                if constexpr (PROJ) {
                    out_f[off] = resid[off] + v;
                } else {
                    out_h[off] = (_Float16)v;
                }
            }
        }
    }
}

// ---------------------------------------------------------------------------
// Kernel 4: per-pixel attention over heads (seq len = 8, d = 64).
// qkv layout [b][o][hw], o = head*192 + {q:0..63, k:64..127, v:128..191}.
// One thread per pixel; reads lane-coalesced; writes abuf TRANSPOSED
// [b][hw][c] in aligned 8B chunks (lane-private contiguous 1KB row).
// ---------------------------------------------------------------------------
__global__ __launch_bounds__(256) void attn_kernel(const _Float16* __restrict__ qkv,
                                                   _Float16* __restrict__ abuf_t) {
    const int p = blockIdx.x * 256 + threadIdx.x;   // 0..4095
    const int b = blockIdx.y;
    const _Float16* base = qkv + (size_t)b * (3 * BDIM_C) * BDIM_HW + p;
    _Float16* outp = abuf_t + ((size_t)b * BDIM_HW + p) * BDIM_C;

    float s[NHEADS][NHEADS];
#pragma unroll
    for (int i = 0; i < NHEADS; ++i)
#pragma unroll
        for (int j = 0; j < NHEADS; ++j) s[i][j] = 0.f;

    for (int dk = 0; dk < DHEAD; ++dk) {
        float qv[NHEADS], kv[NHEADS];
#pragma unroll
        for (int i = 0; i < NHEADS; ++i)
            qv[i] = (float)base[(size_t)(i * 192 + dk) * BDIM_HW];
#pragma unroll
        for (int j = 0; j < NHEADS; ++j)
            kv[j] = (float)base[(size_t)(j * 192 + DHEAD + dk) * BDIM_HW];
#pragma unroll
        for (int i = 0; i < NHEADS; ++i)
#pragma unroll
            for (int j = 0; j < NHEADS; ++j)
                s[i][j] += qv[i] * kv[j];
    }

    const float sc = 0.125f * 1.44269504088896f;    // (1/sqrt(64)) * log2(e)
#pragma unroll
    for (int i = 0; i < NHEADS; ++i) {
        float m = s[i][0];
#pragma unroll
        for (int j = 1; j < NHEADS; ++j) m = fmaxf(m, s[i][j]);
        float l = 0.f;
#pragma unroll
        for (int j = 0; j < NHEADS; ++j) {
            float e = exp2f((s[i][j] - m) * sc);
            s[i][j] = e;
            l += e;
        }
        float inv = 1.f / l;
#pragma unroll
        for (int j = 0; j < NHEADS; ++j) s[i][j] *= inv;
    }

    // A = P V, dk processed in chunks of 4 so each head's output is an
    // aligned 8B vector store into the thread's contiguous output row.
    for (int dk0 = 0; dk0 < DHEAD; dk0 += 4) {
        float vv[NHEADS][4];
#pragma unroll
        for (int j = 0; j < NHEADS; ++j)
#pragma unroll
            for (int e = 0; e < 4; ++e)
                vv[j][e] = (float)base[(size_t)(j * 192 + 2 * DHEAD + dk0 + e) * BDIM_HW];
#pragma unroll
        for (int i = 0; i < NHEADS; ++i) {
            v4h av;
#pragma unroll
            for (int e = 0; e < 4; ++e) {
                float a = 0.f;
#pragma unroll
                for (int j = 0; j < NHEADS; ++j) a += s[i][j] * vv[j][e];
                av[e] = (_Float16)a;
            }
            *(v4h*)(outp + i * DHEAD + dk0) = av;
        }
    }
}

// ---------------------------------------------------------------------------
extern "C" void kernel_launch(void* const* d_in, const int* in_sizes, int n_in,
                              void* d_out, int out_size, void* d_ws, size_t ws_size,
                              hipStream_t stream) {
    const float* x      = (const float*)d_in[0];
    const float* gamma  = (const float*)d_in[1];
    const float* beta   = (const float*)d_in[2];
    const float* w_qkv  = (const float*)d_in[3];
    const float* b_qkv  = (const float*)d_in[4];
    const float* w_proj = (const float*)d_in[5];
    const float* b_proj = (const float*)d_in[6];
    float* out = (float*)d_out;

    char* ws = (char*)d_ws;
    const size_t SZ_XN  = (size_t)BDIM_B * BDIM_C * BDIM_HW * 2;        // 64 MiB
    const size_t SZ_QKV = (size_t)BDIM_B * 3 * BDIM_C * BDIM_HW * 2;    // 192 MiB
    const size_t SZ_A   = SZ_XN;                                        // 64 MiB
    _Float16* xn_t   = (_Float16*)(ws);                                 // [b][hw][c]
    _Float16* qkvh   = (_Float16*)(ws + SZ_XN);                         // [b][o][hw]
    _Float16* abuf_t = (_Float16*)(ws + SZ_XN + SZ_QKV);                // [b][hw][c]
    _Float16* wqh    = (_Float16*)(ws + SZ_XN + SZ_QKV + SZ_A);
    _Float16* wph    = (_Float16*)(ws + SZ_XN + SZ_QKV + SZ_A + (size_t)3 * BDIM_C * BDIM_C * 2);

    // 1) GroupNorm -> transposed f16 activations
    gn_kernel<<<dim3(NGROUPS, BDIM_B), 256, 0, stream>>>(x, gamma, beta, xn_t);

    // 2) weights -> f16
    cvt_kernel<<<dim3(512), 256, 0, stream>>>(w_qkv, wqh, 3 * BDIM_C * BDIM_C);
    cvt_kernel<<<dim3(256), 256, 0, stream>>>(w_proj, wph, BDIM_C * BDIM_C);

    // 3) QKV GEMM: M=1536, N=4096, K=512 per batch
    gemm_kernel<false><<<dim3(GN / 128, (3 * BDIM_C) / 128, BDIM_B), 256, 0, stream>>>(
        wqh, xn_t, b_qkv, nullptr, qkvh, nullptr, 3 * BDIM_C);

    // 4) per-pixel head attention -> transposed output
    attn_kernel<<<dim3(BDIM_HW / 256, BDIM_B), 256, 0, stream>>>(qkvh, abuf_t);

    // 5) proj GEMM + bias + residual: M=512, N=4096, K=512 per batch
    gemm_kernel<true><<<dim3(GN / 128, BDIM_C / 128, BDIM_B), 256, 0, stream>>>(
        wph, abuf_t, b_proj, x, nullptr, out, BDIM_C);
}
